// rgcn_33320356282758
// MI455X (gfx1250) — compile-verified, matrix operand order checked by reading, not run on hardware
//
#include <hip/hip_runtime.h>
#include <stdint.h>

// Problem constants (match reference)
#define N_NODES 4096
#define T_STEPS 365
#define F_IN    16
#define H_DIM   20
#define H4      80
#define K_DIM   4096   // reduction dim of A @ X
#define XT_COLS 48     // 20 (P) + 20 (Q) + 8 zero pad -> 3 tiles of 16

// GEMM tiling
#define KC      1024          // K chunk staged in LDS
#define XPAD    8             // pad elements per column: stride 2064 B (16B-aligned,
                              // dword stride 516 == 4 mod 64 -> conflict-free ds_load_b128)
#define XSTRIDE (KC + XPAD)

typedef __attribute__((ext_vector_type(16))) __bf16 v16bf;
typedef __attribute__((ext_vector_type(8)))  float  v8f;

__device__ __forceinline__ uint16_t f32_to_bf16_bits(float f) {
    uint32_t u = __float_as_uint(f);
    uint32_t r = u + 0x7FFFu + ((u >> 16) & 1u);   // round-to-nearest-even
    return (uint16_t)(r >> 16);
}
__device__ __forceinline__ float sigm(float x) { return 1.0f / (1.0f + __expf(-x)); }

// ---------------------------------------------------------------------------
// Convert A (fp32, 4096x4096 row-major) -> bf16 once per launch.
__global__ void k_convertA(const float* __restrict__ A, uint16_t* __restrict__ Abf) {
    size_t i = ((size_t)blockIdx.x * blockDim.x + threadIdx.x) * 16;
#pragma unroll
    for (int j = 0; j < 16; ++j) Abf[i + j] = f32_to_bf16_bits(A[i + j]);
}

// Zero the 8 pad columns of Xt (cols 40..47), column-major [XT_COLS][K_DIM].
__global__ void k_pad_init(uint16_t* __restrict__ Xt) {
    int i = blockIdx.x * blockDim.x + threadIdx.x;   // 0 .. 8*4096-1
    Xt[(size_t)(40) * K_DIM + i] = 0;
}

// ---------------------------------------------------------------------------
// Warm-up: full-sequence LSTM from zero state (per-node independent).
__global__ void k_warmup(const float* __restrict__ x_all,
                         const float* __restrict__ Wk, const float* __restrict__ Wr,
                         const float* __restrict__ bias,
                         float* __restrict__ h_out, float* __restrict__ c_out) {
    __shared__ float sWk[F_IN * H4], sWr[H_DIM * H4], sB[H4];
    for (int i = threadIdx.x; i < F_IN * H4;  i += blockDim.x) sWk[i] = Wk[i];
    for (int i = threadIdx.x; i < H_DIM * H4; i += blockDim.x) sWr[i] = Wr[i];
    for (int i = threadIdx.x; i < H4;         i += blockDim.x) sB[i]  = bias[i];
    __syncthreads();

    int node = blockIdx.x * blockDim.x + threadIdx.x;
    if (node >= N_NODES) return;

    float h[H_DIM], c[H_DIM];
#pragma unroll
    for (int j = 0; j < H_DIM; ++j) { h[j] = 0.f; c[j] = 0.f; }

    const float* xp = x_all + (size_t)node * T_STEPS * F_IN;
    for (int t = 0; t < T_STEPS; ++t) {
        float x[F_IN];
#pragma unroll
        for (int k = 0; k < F_IN; ++k) x[k] = xp[t * F_IN + k];
        float hn[H_DIM], cn[H_DIM];
#pragma unroll
        for (int j = 0; j < H_DIM; ++j) {
            float zi = sB[j], zf = sB[H_DIM + j], zg = sB[2 * H_DIM + j], zo = sB[3 * H_DIM + j];
#pragma unroll
            for (int k = 0; k < F_IN; ++k) {
                float xv = x[k]; const float* w = &sWk[k * H4];
                zi += xv * w[j]; zf += xv * w[H_DIM + j];
                zg += xv * w[2 * H_DIM + j]; zo += xv * w[3 * H_DIM + j];
            }
#pragma unroll
            for (int k = 0; k < H_DIM; ++k) {
                float hv = h[k]; const float* w = &sWr[k * H4];
                zi += hv * w[j]; zf += hv * w[H_DIM + j];
                zg += hv * w[2 * H_DIM + j]; zo += hv * w[3 * H_DIM + j];
            }
            float cv = sigm(zf) * c[j] + sigm(zi) * tanhf(zg);
            cn[j] = cv; hn[j] = sigm(zo) * tanhf(cv);
        }
#pragma unroll
        for (int j = 0; j < H_DIM; ++j) { h[j] = hn[j]; c[j] = cn[j]; }
    }
#pragma unroll
    for (int j = 0; j < H_DIM; ++j) {
        h_out[node * H_DIM + j] = h[j];
        c_out[node * H_DIM + j] = c[j];
    }
}

// ---------------------------------------------------------------------------
// Per-step node kernel (pre-graph): P = h@Wgh+bgh, Q = c@Wgc+bgc written as
// bf16 into column-major Xt [48][4096]; also the LSTM cell -> h_cur, c_cur.
__global__ void k_node_pre(const float* __restrict__ x_all,
                           const float* __restrict__ Wk, const float* __restrict__ Wr,
                           const float* __restrict__ bias,
                           const float* __restrict__ Wgh, const float* __restrict__ bgh,
                           const float* __restrict__ Wgc, const float* __restrict__ bgc,
                           const float* __restrict__ h_prev, const float* __restrict__ c_prev,
                           uint16_t* __restrict__ Xt,
                           float* __restrict__ h_cur, float* __restrict__ c_cur, int t) {
    __shared__ float sWk[F_IN * H4], sWr[H_DIM * H4], sB[H4];
    __shared__ float sWgh[H_DIM * H_DIM], sWgc[H_DIM * H_DIM], sbgh[H_DIM], sbgc[H_DIM];
    for (int i = threadIdx.x; i < F_IN * H4;      i += blockDim.x) sWk[i]  = Wk[i];
    for (int i = threadIdx.x; i < H_DIM * H4;     i += blockDim.x) sWr[i]  = Wr[i];
    for (int i = threadIdx.x; i < H4;             i += blockDim.x) sB[i]   = bias[i];
    for (int i = threadIdx.x; i < H_DIM * H_DIM;  i += blockDim.x) { sWgh[i] = Wgh[i]; sWgc[i] = Wgc[i]; }
    for (int i = threadIdx.x; i < H_DIM;          i += blockDim.x) { sbgh[i] = bgh[i]; sbgc[i] = bgc[i]; }
    __syncthreads();

    int node = blockIdx.x * blockDim.x + threadIdx.x;
    if (node >= N_NODES) return;

    float h[H_DIM], c[H_DIM];
#pragma unroll
    for (int j = 0; j < H_DIM; ++j) {
        h[j] = h_prev[node * H_DIM + j];
        c[j] = c_prev[node * H_DIM + j];
    }

    // P = h@Wgh + bgh ; Q = c@Wgc + bgc  -> bf16, column-major Xt
#pragma unroll
    for (int j = 0; j < H_DIM; ++j) {
        float p = sbgh[j], q = sbgc[j];
#pragma unroll
        for (int k = 0; k < H_DIM; ++k) {
            p += h[k] * sWgh[k * H_DIM + j];
            q += c[k] * sWgc[k * H_DIM + j];
        }
        Xt[(size_t)j * K_DIM + node]           = f32_to_bf16_bits(p);
        Xt[(size_t)(H_DIM + j) * K_DIM + node] = f32_to_bf16_bits(q);
    }

    // LSTM cell with x_t
    float x[F_IN];
    const float* xp = x_all + (size_t)node * T_STEPS * F_IN + (size_t)t * F_IN;
#pragma unroll
    for (int k = 0; k < F_IN; ++k) x[k] = xp[k];
#pragma unroll
    for (int j = 0; j < H_DIM; ++j) {
        float zi = sB[j], zf = sB[H_DIM + j], zg = sB[2 * H_DIM + j], zo = sB[3 * H_DIM + j];
#pragma unroll
        for (int k = 0; k < F_IN; ++k) {
            float xv = x[k]; const float* w = &sWk[k * H4];
            zi += xv * w[j]; zf += xv * w[H_DIM + j];
            zg += xv * w[2 * H_DIM + j]; zo += xv * w[3 * H_DIM + j];
        }
#pragma unroll
        for (int k = 0; k < H_DIM; ++k) {
            float hv = h[k]; const float* w = &sWr[k * H4];
            zi += hv * w[j]; zf += hv * w[H_DIM + j];
            zg += hv * w[2 * H_DIM + j]; zo += hv * w[3 * H_DIM + j];
        }
        float cv = sigm(zf) * c[j] + sigm(zi) * tanhf(zg);
        c_cur[node * H_DIM + j] = cv;
        h_cur[node * H_DIM + j] = sigm(zo) * tanhf(cv);
    }
}

// ---------------------------------------------------------------------------
// WMMA GEMM: G = tanh(A_bf16[4096x4096] @ X[4096x48]).
//  - X staged through LDS in K-chunks of 1024 (97 KB, conflict-free padded
//    stride): B global traffic drops from 96 MB/step to 6 MB/step.
//  - Each wave owns 2 row tiles (32 rows) x 3 col tiles: B fragments feed
//    6 WMMAs. Block = 8 waves = 256 rows; grid = 16 blocks = 4096 rows.
//  - A (32 MB bf16, L2-resident) is streamed once per step with prefetch.
__global__ void __launch_bounds__(256) k_gemm(const uint16_t* __restrict__ Abf,
                                              const uint16_t* __restrict__ Xt,
                                              float* __restrict__ G) {
    __shared__ uint16_t sX[XT_COLS * XSTRIDE];   // 48 * 1032 * 2 = 99,072 B

    int lane   = threadIdx.x & 31;
    int wave   = threadIdx.x >> 5;
    int laneN  = lane & 15;
    int laneHi = lane >> 4;

    int rowBase0 = (blockIdx.x * 16 + wave * 2) * 16;   // first 16-row tile
    int rowBase1 = rowBase0 + 16;                       // second 16-row tile

    // 16-bit A fragment (16x32): lane = M row, VGPR v packs a K pair
    //   K(v, laneHi) = (v<4 ? 2v : 16+2(v-4)) + 8*laneHi
    // => two contiguous 16-byte loads at byte offsets laneHi*16 and +32.
    const uint16_t* arow0 = Abf + (size_t)(rowBase0 + laneN) * K_DIM;
    const uint16_t* arow1 = Abf + (size_t)(rowBase1 + laneN) * K_DIM;

    union Frag { v16bf v; uint4 u[2]; };
    v8f acc[2][3] = {};

    for (int kb = 0; kb < K_DIM; kb += KC) {
        __syncthreads();   // previous chunk fully consumed
        // Cooperative stage: 48 cols x 1024 elems = 6144 uint4, 24 per thread.
        for (int u = threadIdx.x; u < XT_COLS * (KC / 8); u += 256) {
            int col = u >> 7;        // u / 128
            int k8  = u & 127;       // u % 128
            uint4 v = *(const uint4*)(Xt + (size_t)col * K_DIM + kb + k8 * 8);
            *(uint4*)(sX + col * XSTRIDE + k8 * 8) = v;
        }
        __syncthreads();

        for (int kl = 0; kl < KC; kl += 32) {
            int off  = kb + kl + laneHi * 8;   // global K offset (elements)
            int loff = kl + laneHi * 8;        // LDS K offset (elements)
            __builtin_prefetch(arow0 + off + 2048, 0, 1);
            Frag a0, a1, b[3];
            a0.u[0] = *(const uint4*)(arow0 + off);
            a0.u[1] = *(const uint4*)(arow0 + off + 16);
            a1.u[0] = *(const uint4*)(arow1 + off);
            a1.u[1] = *(const uint4*)(arow1 + off + 16);
#pragma unroll
            for (int j = 0; j < 3; ++j) {
                const uint16_t* bp = sX + (j * 16 + laneN) * XSTRIDE + loff;
                b[j].u[0] = *(const uint4*)(bp);
                b[j].u[1] = *(const uint4*)(bp + 16);
            }
#pragma unroll
            for (int j = 0; j < 3; ++j) {
                acc[0][j] = __builtin_amdgcn_wmma_f32_16x16x32_bf16(
                    false, a0.v, false, b[j].v, (short)0, acc[0][j], false, false);
                acc[1][j] = __builtin_amdgcn_wmma_f32_16x16x32_bf16(
                    false, a1.v, false, b[j].v, (short)0, acc[1][j], false, false);
            }
        }
    }

    // C/D layout: VGPR r -> M = r + 8*laneHi, N = laneN. Fuse tanh.
#pragma unroll
    for (int rt = 0; rt < 2; ++rt) {
        int rowBase = rt ? rowBase1 : rowBase0;
#pragma unroll
        for (int r = 0; r < 8; ++r) {
            float* g = G + (size_t)(rowBase + r + laneHi * 8) * XT_COLS;
#pragma unroll
            for (int j = 0; j < 3; ++j)
                g[j * 16 + laneN] = tanhf(acc[rt][j][r]);
        }
    }
}

// ---------------------------------------------------------------------------
// Per-step node kernel (post-graph): update gates + output row.
__global__ void k_node_post(const float* __restrict__ G,
                            const float* __restrict__ h_cur, const float* __restrict__ c_cur,
                            const float* __restrict__ Whc, const float* __restrict__ Whp,
                            const float* __restrict__ bh,
                            const float* __restrict__ Wcc, const float* __restrict__ Wcp,
                            const float* __restrict__ bc,
                            const float* __restrict__ Wout, const float* __restrict__ bout,
                            float* __restrict__ h_prev, float* __restrict__ c_prev,
                            float* __restrict__ out, int t) {
    __shared__ float sWhc[H_DIM * H_DIM], sWhp[H_DIM * H_DIM];
    __shared__ float sWcc[H_DIM * H_DIM], sWcp[H_DIM * H_DIM];
    __shared__ float sbh[H_DIM], sbc[H_DIM], sWo[H_DIM], sbo;
    for (int i = threadIdx.x; i < H_DIM * H_DIM; i += blockDim.x) {
        sWhc[i] = Whc[i]; sWhp[i] = Whp[i]; sWcc[i] = Wcc[i]; sWcp[i] = Wcp[i];
    }
    for (int i = threadIdx.x; i < H_DIM; i += blockDim.x) {
        sbh[i] = bh[i]; sbc[i] = bc[i]; sWo[i] = Wout[i];
    }
    if (threadIdx.x == 0) sbo = bout[0];
    __syncthreads();

    int node = blockIdx.x * blockDim.x + threadIdx.x;
    if (node >= N_NODES) return;

    float hc[H_DIM], cc[H_DIM], hg[H_DIM], cg[H_DIM];
    const float* grow = G + (size_t)node * XT_COLS;
#pragma unroll
    for (int j = 0; j < H_DIM; ++j) {
        hc[j] = h_cur[node * H_DIM + j];
        cc[j] = c_cur[node * H_DIM + j];
        hg[j] = grow[j];
        cg[j] = grow[H_DIM + j];
    }
    float o = sbo;
#pragma unroll
    for (int j = 0; j < H_DIM; ++j) {
        float zh = sbh[j], zc = sbc[j];
#pragma unroll
        for (int k = 0; k < H_DIM; ++k) {
            zh += hc[k] * sWhc[k * H_DIM + j] + hg[k] * sWhp[k * H_DIM + j];
            zc += cc[k] * sWcc[k * H_DIM + j] + cg[k] * sWcp[k * H_DIM + j];
        }
        float hv = sigm(zh), cv = sigm(zc);
        h_prev[node * H_DIM + j] = hv;
        c_prev[node * H_DIM + j] = cv;
        o += hv * sWo[j];
    }
    out[(size_t)(t - 1) * N_NODES + node] = o;
}

// ---------------------------------------------------------------------------
// Workspace layout (bytes)
#define OFF_ABF   ((size_t)0)
#define SZ_ABF    ((size_t)K_DIM * K_DIM * 2)               // 33,554,432
#define OFF_XT    (OFF_ABF + SZ_ABF)
#define SZ_XT     ((size_t)XT_COLS * K_DIM * 2)             // 393,216
#define OFF_G     (OFF_XT + SZ_XT)
#define SZ_G      ((size_t)N_NODES * XT_COLS * 4)           // 786,432
#define OFF_HPREV (OFF_G + SZ_G)
#define SZ_HC     ((size_t)N_NODES * H_DIM * 4)             // 327,680
#define OFF_CPREV (OFF_HPREV + SZ_HC)
#define OFF_HCUR  (OFF_CPREV + SZ_HC)
#define OFF_CCUR  (OFF_HCUR + SZ_HC)
#define WS_NEED   (OFF_CCUR + SZ_HC)

extern "C" void kernel_launch(void* const* d_in, const int* in_sizes, int n_in,
                              void* d_out, int out_size, void* d_ws, size_t ws_size,
                              hipStream_t stream) {
    const float* inputs = (const float*)d_in[0];
    const float* A      = (const float*)d_in[1];
    const float* Wk     = (const float*)d_in[2];
    const float* Wr     = (const float*)d_in[3];
    const float* bias   = (const float*)d_in[4];
    const float* Wgh    = (const float*)d_in[5];
    const float* bgh    = (const float*)d_in[6];
    const float* Wgc    = (const float*)d_in[7];
    const float* bgc    = (const float*)d_in[8];
    const float* Whc    = (const float*)d_in[9];
    const float* Whp    = (const float*)d_in[10];
    const float* bh     = (const float*)d_in[11];
    const float* Wcc    = (const float*)d_in[12];
    const float* Wcp    = (const float*)d_in[13];
    const float* bc     = (const float*)d_in[14];
    const float* Wout   = (const float*)d_in[15];
    const float* bout   = (const float*)d_in[16];
    float* out = (float*)d_out;

    if (ws_size < WS_NEED) return;
    uint8_t* ws = (uint8_t*)d_ws;
    uint16_t* Abf   = (uint16_t*)(ws + OFF_ABF);
    uint16_t* Xt    = (uint16_t*)(ws + OFF_XT);
    float*    G      = (float*)(ws + OFF_G);
    float*    h_prev = (float*)(ws + OFF_HPREV);
    float*    c_prev = (float*)(ws + OFF_CPREV);
    float*    h_cur  = (float*)(ws + OFF_HCUR);
    float*    c_cur  = (float*)(ws + OFF_CCUR);

    // A -> bf16 (resident in 192MB L2 for all 364 steps)
    k_convertA<<<4096, 256, 0, stream>>>(A, Abf);
    k_pad_init<<<128, 256, 0, stream>>>(Xt);
    // Warm-up LSTM over full sequence
    k_warmup<<<16, 256, 0, stream>>>(inputs, Wk, Wr, bias, h_prev, c_prev);

    for (int t = 1; t < T_STEPS; ++t) {
        k_node_pre<<<16, 256, 0, stream>>>(inputs, Wk, Wr, bias, Wgh, bgh, Wgc, bgc,
                                           h_prev, c_prev, Xt, h_cur, c_cur, t);
        k_gemm<<<16, 256, 0, stream>>>(Abf, Xt, G);
        k_node_post<<<16, 256, 0, stream>>>(G, h_cur, c_cur, Whc, Whp, bh, Wcc, Wcp, bc,
                                            Wout, bout, h_prev, c_prev, out, t);
    }
}
